// SingleActionRelGraph_1958505087381
// MI455X (gfx1250) — matrix operand order repaired and checked by reading, not compiled, not run on hardware
//
#include <hip/hip_runtime.h>
#include <cstdint>
#include <cstddef>

typedef __attribute__((ext_vector_type(16))) __bf16 v16bf;
typedef __attribute__((ext_vector_type(8)))  float  v8f;

namespace {

constexpr int HID = 128;   // HIDDEN
constexpr int HD  = 512;   // HEADS * HEAD_DIM
constexpr int KTILES = 4;  // K=128 -> 4 x 32

__device__ __forceinline__ unsigned short f2bf(float f) {
  unsigned u = __float_as_uint(f);
  u += 0x7FFFu + ((u >> 16) & 1u);          // round-to-nearest-even
  return (unsigned short)(u >> 16);
}
__device__ __forceinline__ unsigned fkey(float f) {
  unsigned u = __float_as_uint(f);
  return (u & 0x80000000u) ? ~u : (u | 0x80000000u);
}
__device__ __forceinline__ float keyToF(unsigned k) {
  unsigned u = (k & 0x80000000u) ? (k & 0x7FFFFFFFu) : ~k;
  return __uint_as_float(u);
}

// ---------- pack activations into WMMA A-fragment layout (16x32 bf16) ----------
__global__ void pack_x_kernel(const float* __restrict__ x, unsigned short* __restrict__ pA,
                              int Nrows, int RT) {
  int idx = blockIdx.x * blockDim.x + threadIdx.x;
  if (idx >= RT * 2048) return;
  int j    = idx & 15;
  int lane = (idx >> 4) & 31;
  int kt   = (idx >> 9) & 3;
  int rt   = idx >> 11;
  int m = rt * 16 + (lane & 15);
  int kbase = (lane < 16) ? 0 : 8;
  int k = kt * 32 + ((j < 8) ? (kbase + j) : (16 + kbase + (j - 8)));
  float v = (m < Nrows) ? x[(size_t)m * HID + k] : 0.0f;
  pA[idx] = f2bf(v);
}

// ---------- pack weights [128 x Ncols] into WMMA B-fragment layout (32x16 bf16) ----------
__global__ void pack_w_kernel(const float* __restrict__ W, unsigned short* __restrict__ pB,
                              int Ncols, int NT) {
  int idx = blockIdx.x * blockDim.x + threadIdx.x;
  if (idx >= KTILES * NT * 512) return;
  int j    = idx & 15;
  int lane = (idx >> 4) & 31;
  int frag = idx >> 9;
  int nt = frag % NT;
  int kt = frag / NT;
  int n = nt * 16 + (lane & 15);
  int k = kt * 32 + ((j < 8) ? (((lane < 16) ? 0 : 8) + j) : (16 + ((lane < 16) ? 0 : 8) + (j - 8)));
  float v = (n < Ncols) ? W[(size_t)k * Ncols + n] : 0.0f;
  pB[idx] = f2bf(v);
}

// ---------- WMMA GEMM: C[M x Ncols] = A[M x 128] @ B[128 x Ncols] + bias ----------
// one wave per 16x64 output tile; A-fragment reused across 4 N-tiles
__global__ void wmma_gemm_kernel(const unsigned short* __restrict__ pA,
                                 const unsigned short* __restrict__ pB,
                                 const float* __restrict__ bias,
                                 float* __restrict__ out,
                                 int M, int Ncols, int NT, int RT) {
  const int lane = threadIdx.x & 31;
  const int gw   = (blockIdx.x * blockDim.x + threadIdx.x) >> 5;
  const int ng   = NT >> 2;
  if (gw >= RT * ng) return;
  const int rt = gw / ng;
  const int g  = gw % ng;

  const v16bf* A = reinterpret_cast<const v16bf*>(pA);
  const v16bf* B = reinterpret_cast<const v16bf*>(pB);

  v8f acc[4] = {v8f{}, v8f{}, v8f{}, v8f{}};
#pragma unroll
  for (int kt = 0; kt < KTILES; ++kt) {
    v16bf a = A[((size_t)rt * KTILES + kt) * 32 + lane];
#pragma unroll
    for (int t = 0; t < 4; ++t) {
      v16bf b = B[((size_t)kt * NT + (g * 4 + t)) * 32 + lane];
      acc[t] = __builtin_amdgcn_wmma_f32_16x16x32_bf16(false, a, false, b,
                                                       (short)0, acc[t], false, false);
    }
  }

  const int rowBase = rt * 16 + ((lane >= 16) ? 8 : 0);
  const int c0 = lane & 15;
#pragma unroll
  for (int t = 0; t < 4; ++t) {
    int col = (g * 4 + t) * 16 + c0;
    if (col < Ncols) {
      float bv = bias ? bias[col] : 0.0f;
#pragma unroll
      for (int i = 0; i < 8; ++i) {
        int row = rowBase + i;
        if (row < M) out[(size_t)row * Ncols + col] = acc[t][i] + bv;
      }
    }
  }
}

// ---------- initial action embedding ----------
__global__ void embed_kernel(const float* __restrict__ ax, const int* __restrict__ tix,
                             const int* __restrict__ bix,
                             const float* __restrict__ W, const float* __restrict__ b,
                             const float* __restrict__ temb, const float* __restrict__ bemb,
                             float* __restrict__ h, int NA) {
  int i = blockIdx.x * blockDim.x + threadIdx.x;
  if (i >= NA * HID) return;
  int n = i >> 7, d = i & 127;
  const float* xr = ax + (size_t)n * 7;
  float acc = b[d];
#pragma unroll
  for (int j = 0; j < 7; ++j) acc += xr[j] * W[j * HID + d];
  acc += temb[(size_t)tix[n] * HID + d] + bemb[(size_t)bix[n] * HID + d];
  h[i] = acc;
}

__global__ void copy_f32_kernel(const float* __restrict__ s, float* __restrict__ d, long n) {
  long i = (long)blockIdx.x * blockDim.x + threadIdx.x;
  if (i < n) d[i] = s[i];
}
__global__ void zero_f32_kernel(float* __restrict__ p, long n) {
  long i = (long)blockIdx.x * blockDim.x + threadIdx.x;
  if (i < n) p[i] = 0.0f;
}
__global__ void fill_u32_kernel(unsigned* __restrict__ p, unsigned v, long n) {
  long i = (long)blockIdx.x * blockDim.x + threadIdx.x;
  if (i < n) p[i] = v;
}

// ---------- edge logits + segment max (one wave per (edge, head)) ----------
__global__ void attn_logits_kernel(const float* __restrict__ Q, const float* __restrict__ Kt,
                                   const int* __restrict__ src, const int* __restrict__ dst,
                                   const float* __restrict__ ea, const float* __restrict__ We,
                                   float* __restrict__ logits, unsigned* __restrict__ mkey,
                                   int E) {
  int lane = threadIdx.x & 31;
  int gw = blockIdx.x * (blockDim.x >> 5) + (threadIdx.x >> 5);
  if (gw >= E * 4) return;
  int e = gw >> 2, h = gw & 3;
  int s = src[e], d = dst[e];
  float ea0 = 0, ea1 = 0, ea2 = 0, ea3 = 0;
  if (ea) { ea0 = ea[e * 4 + 0]; ea1 = ea[e * 4 + 1]; ea2 = ea[e * 4 + 2]; ea3 = ea[e * 4 + 3]; }
  float dot = 0.0f;
#pragma unroll
  for (int t = 0; t < 4; ++t) {
    int dd = h * 128 + lane + 32 * t;
    float kv = Kt[(size_t)s * HD + dd];
    if (ea) kv += ea0 * We[dd] + ea1 * We[512 + dd] + ea2 * We[1024 + dd] + ea3 * We[1536 + dd];
    dot += Q[(size_t)d * HD + dd] * kv;
  }
#pragma unroll
  for (int off = 16; off > 0; off >>= 1) dot += __shfl_xor(dot, off, 32);
  if (lane == 0) {
    float lg = dot * 0.08838834764831845f;   // 1/sqrt(128)
    logits[gw] = lg;
    atomicMax(&mkey[(size_t)d * 4 + h], fkey(lg));
  }
}

// ---------- exp + segment sum ----------
__global__ void expden_kernel(const float* __restrict__ logits, const int* __restrict__ dst,
                              const unsigned* __restrict__ mkey, float* __restrict__ exb,
                              float* __restrict__ den, int E) {
  int i = blockIdx.x * blockDim.x + threadIdx.x;
  if (i >= E * 4) return;
  int e = i >> 2, h = i & 3;
  int d = dst[e];
  float m = keyToF(mkey[(size_t)d * 4 + h]);
  float ex = __expf(logits[i] - m);
  exb[i] = ex;
  atomicAdd(&den[(size_t)d * 4 + h], ex);
}

// ---------- alpha * (v + e) scatter-accumulate ----------
__global__ void attn_accum_kernel(const float* __restrict__ V,
                                  const int* __restrict__ src, const int* __restrict__ dst,
                                  const float* __restrict__ ea, const float* __restrict__ We,
                                  const float* __restrict__ exb, const float* __restrict__ den,
                                  float* __restrict__ agg, int E) {
  int lane = threadIdx.x & 31;
  int gw = blockIdx.x * (blockDim.x >> 5) + (threadIdx.x >> 5);
  if (gw >= E * 4) return;
  int e = gw >> 2, h = gw & 3;
  int s = src[e], d = dst[e];
  float alpha = exb[gw] / den[(size_t)d * 4 + h];
  float ea0 = 0, ea1 = 0, ea2 = 0, ea3 = 0;
  if (ea) { ea0 = ea[e * 4 + 0]; ea1 = ea[e * 4 + 1]; ea2 = ea[e * 4 + 2]; ea3 = ea[e * 4 + 3]; }
#pragma unroll
  for (int t = 0; t < 4; ++t) {
    int dd = h * 128 + lane + 32 * t;
    float vv = V[(size_t)s * HD + dd];
    if (ea) vv += ea0 * We[dd] + ea1 * We[512 + dd] + ea2 * We[1024 + dd] + ea3 * We[1536 + dd];
    atomicAdd(&agg[(size_t)d * HD + dd], alpha * vv);
  }
}

// ---------- mean-over-heads + skip (+pre) (+relu) ----------
__global__ void combine_kernel(const float* __restrict__ agg, const float* __restrict__ skip,
                               const float* __restrict__ pre, float* __restrict__ out,
                               int N, int doRelu) {
  int i = blockIdx.x * blockDim.x + threadIdx.x;
  if (i >= N * HID) return;
  int n = i >> 7, d = i & 127;
  const float* a = agg + (size_t)n * HD;
  float v = 0.25f * (a[d] + a[128 + d] + a[256 + d] + a[384 + d]) + skip[i];
  if (pre) v += pre[i];
  if (doRelu) v = fmaxf(v, 0.0f);
  out[i] = v;
}

} // namespace

extern "C" void kernel_launch(void* const* d_in, const int* in_sizes, int n_in,
                              void* d_out, int out_size, void* d_ws, size_t ws_size,
                              hipStream_t stream) {
  (void)n_in; (void)out_size; (void)ws_size;

  const float* action_x = (const float*)d_in[0];
  const int*   type_idx = (const int*)d_in[1];
  const int*   body_idx = (const int*)d_in[2];
  const int*   ei_aa    = (const int*)d_in[3];
  const float* ea_aa    = (const float*)d_in[4];
  const int*   ei_pa    = (const int*)d_in[5];
  const int*   ei_ap    = (const int*)d_in[6];

  const int NA = in_sizes[0] / 7;
  const int E_aa = in_sizes[3] / 2, E_pa = in_sizes[5] / 2, E_ap = in_sizes[6] / 2;
  const int NP = in_sizes[11] / HID;
  const int RT_A = (NA + 15) / 16;
  const int RT_P = (NP + 15) / 16;
  int Emax = E_aa > E_pa ? E_aa : E_pa; if (E_ap > Emax) Emax = E_ap;

  // params flat order: 7..14 top-level, then per (layer, rel): Wq,bq,Wk,bk,Wv,bv,We,Wskip,bskip
  auto convBase = [](int l, int r) { return 15 + (l * 3 + r) * 9; };

  // ---- workspace bump allocator ----
  char* ws = (char*)d_ws;
  size_t off = 0;
  auto alloc = [&](size_t bytes) -> void* {
    void* p = ws + off;
    off = (off + bytes + 255) & ~(size_t)255;
    return p;
  };

  unsigned short* packA = (unsigned short*)alloc((size_t)RT_A * 2048 * 2);
  unsigned short* packP = (unsigned short*)alloc((size_t)RT_P * 2048 * 2);
  unsigned short *pWq[2][3], *pWk[2][3], *pWv[2][3], *pWs[2][3];
  for (int l = 0; l < 2; ++l)
    for (int r = 0; r < 3; ++r) {
      pWq[l][r] = (unsigned short*)alloc(65536 * 2);
      pWk[l][r] = (unsigned short*)alloc(65536 * 2);
      pWv[l][r] = (unsigned short*)alloc(65536 * 2);
      pWs[l][r] = (unsigned short*)alloc(16384 * 2);
    }
  unsigned short* pHead = (unsigned short*)alloc(8192 * 2);

  float* hA0  = (float*)alloc((size_t)NA * HID * 4);
  float* hA1  = (float*)alloc((size_t)NA * HID * 4);
  float* hP0  = (float*)alloc((size_t)NP * HID * 4);
  float* hP1  = (float*)alloc((size_t)NP * HID * 4);
  float* preA = (float*)alloc((size_t)NA * HID * 4);
  float* Qb   = (float*)alloc((size_t)NA * HD * 4);
  float* Kb   = (float*)alloc((size_t)NA * HD * 4);
  float* Vb   = (float*)alloc((size_t)NA * HD * 4);
  float* Sb   = (float*)alloc((size_t)NA * HID * 4);
  float* agg  = (float*)alloc((size_t)NA * HD * 4);
  float* logitsB = (float*)alloc((size_t)Emax * 4 * 4);
  float* exB     = (float*)alloc((size_t)Emax * 4 * 4);
  unsigned* mkey = (unsigned*)alloc((size_t)NA * 4 * 4);
  float* den     = (float*)alloc((size_t)NA * 4 * 4);

  auto cdiv = [](long a, long b) { return (int)((a + b - 1) / b); };

  // ---- pack all weights (bf16 fragment layout) ----
  for (int l = 0; l < 2; ++l)
    for (int r = 0; r < 3; ++r) {
      int base = convBase(l, r);
      pack_w_kernel<<<cdiv(65536, 256), 256, 0, stream>>>((const float*)d_in[base + 0], pWq[l][r], HD, 32);
      pack_w_kernel<<<cdiv(65536, 256), 256, 0, stream>>>((const float*)d_in[base + 2], pWk[l][r], HD, 32);
      pack_w_kernel<<<cdiv(65536, 256), 256, 0, stream>>>((const float*)d_in[base + 4], pWv[l][r], HD, 32);
      pack_w_kernel<<<cdiv(16384, 256), 256, 0, stream>>>((const float*)d_in[base + 7], pWs[l][r], HID, 8);
    }
  pack_w_kernel<<<cdiv(8192, 256), 256, 0, stream>>>((const float*)d_in[13], pHead, 42, 4);

  // ---- initial node features ----
  embed_kernel<<<cdiv((long)NA * HID, 256), 256, 0, stream>>>(
      action_x, type_idx, body_idx, (const float*)d_in[7], (const float*)d_in[8],
      (const float*)d_in[9], (const float*)d_in[10], hA0, NA);
  copy_f32_kernel<<<cdiv((long)NP * HID, 256), 256, 0, stream>>>(
      (const float*)d_in[11], hP0, (long)NP * HID);

  auto gemm = [&](const unsigned short* pA, const unsigned short* pB, const float* bias,
                  float* out, int M, int Ncols, int NT, int RT) {
    int waves = RT * (NT / 4);
    wmma_gemm_kernel<<<cdiv(waves, 4), 128, 0, stream>>>(pA, pB, bias, out, M, Ncols, NT, RT);
  };

  auto runConv = [&](const unsigned short* pDst, int RTd, int Md,
                     const unsigned short* pSrc, int RTs, int Ms,
                     int l, int r, const int* sIdx, const int* dIdx, int E,
                     const float* eattr) {
    int base = convBase(l, r);
    gemm(pDst, pWq[l][r], (const float*)d_in[base + 1], Qb, Md, HD, 32, RTd);
    gemm(pSrc, pWk[l][r], (const float*)d_in[base + 3], Kb, Ms, HD, 32, RTs);
    gemm(pSrc, pWv[l][r], (const float*)d_in[base + 5], Vb, Ms, HD, 32, RTs);
    gemm(pDst, pWs[l][r], (const float*)d_in[base + 8], Sb, Md, HID, 8, RTd);
    zero_f32_kernel<<<cdiv((long)Md * HD, 256), 256, 0, stream>>>(agg, (long)Md * HD);
    fill_u32_kernel<<<cdiv((long)Md * 4, 256), 256, 0, stream>>>(mkey, 0x007FFFFFu, (long)Md * 4);
    zero_f32_kernel<<<cdiv((long)Md * 4, 256), 256, 0, stream>>>(den, (long)Md * 4);
    const float* We = eattr ? (const float*)d_in[base + 6] : nullptr;
    attn_logits_kernel<<<cdiv((long)E * 4, 8), 256, 0, stream>>>(Qb, Kb, sIdx, dIdx, eattr, We, logitsB, mkey, E);
    expden_kernel<<<cdiv((long)E * 4, 256), 256, 0, stream>>>(logitsB, dIdx, mkey, exB, den, E);
    attn_accum_kernel<<<cdiv((long)E * 4, 8), 256, 0, stream>>>(Vb, sIdx, dIdx, eattr, We, exB, den, agg, E);
  };

  float *hA = hA0, *hAn = hA1, *hP = hP0, *hPn = hP1;
  for (int l = 0; l < 2; ++l) {
    pack_x_kernel<<<cdiv((long)RT_A * 2048, 256), 256, 0, stream>>>(hA, packA, NA, RT_A);
    pack_x_kernel<<<cdiv((long)RT_P * 2048, 256), 256, 0, stream>>>(hP, packP, NP, RT_P);

    // action<-action (has edge attrs)
    runConv(packA, RT_A, NA, packA, RT_A, NA, l, 0, ei_aa, ei_aa + E_aa, E_aa, ea_aa);
    combine_kernel<<<cdiv((long)NA * HID, 256), 256, 0, stream>>>(agg, Sb, nullptr, preA, NA, 0);

    // action<-player
    runConv(packA, RT_A, NA, packP, RT_P, NP, l, 1, ei_pa, ei_pa + E_pa, E_pa, nullptr);
    combine_kernel<<<cdiv((long)NA * HID, 256), 256, 0, stream>>>(agg, Sb, preA, hAn, NA, 1);

    // player<-action
    runConv(packP, RT_P, NP, packA, RT_A, NA, l, 2, ei_ap, ei_ap + E_ap, E_ap, nullptr);
    combine_kernel<<<cdiv((long)NP * HID, 256), 256, 0, stream>>>(agg, Sb, nullptr, hPn, NP, 1);

    float* t;
    t = hA; hA = hAn; hAn = t;
    t = hP; hP = hPn; hPn = t;
  }

  // ---- output head: [NA x 128] @ [128 x 42] + b ----
  pack_x_kernel<<<cdiv((long)RT_A * 2048, 256), 256, 0, stream>>>(hA, packA, NA, RT_A);
  gemm(packA, pHead, (const float*)d_in[14], (float*)d_out, NA, 42, 4, RT_A);
}
